// MLAAttention_58119497449609
// MI455X (gfx1250) — matrix operand order repaired and checked by reading, not compiled, not run on hardware
//
#include <hip/hip_runtime.h>

typedef __attribute__((ext_vector_type(16))) _Float16 v16h;
typedef __attribute__((ext_vector_type(8)))  _Float16 v8h;
typedef __attribute__((ext_vector_type(8)))  float    v8f;
typedef __attribute__((ext_vector_type(4)))  float    v4f;

#define GROUPS   32     // B*H
#define LSEQ     2048
#define DH       128
#define BM       128    // query rows per block (8 waves x 16)
#define BN       64     // keys per iteration
#define NHEADS   16
#define SK_STR   136    // padded half-stride for K tile rows   (mult of 8)
#define SV_STR   72     // padded half-stride for V^T tile rows (mult of 8)
#define SP_STR   72     // padded half-stride for P rows        (mult of 8)

__global__ __launch_bounds__(256) void fa_fwd_f32_wmma(
    const float* __restrict__ q, const float* __restrict__ k,
    const float* __restrict__ v, float* __restrict__ out)
{
    __shared__ _Float16 sK [BN * SK_STR];        // K tile,  row-major [n][d]
    __shared__ _Float16 sVt[DH * SV_STR];        // V tile,  transposed [d][n]
    __shared__ _Float16 sP [8][16 * SP_STR];     // per-wave P bounce  [m][n]

    const int tid    = threadIdx.x;
    const int w      = tid >> 5;
    const int lane   = tid & 31;
    const int lanelo = lane & 15;
    const int khalf  = lane >> 4;                // 0: lanes 0-15, 1: lanes 16-31
    const int mb     = blockIdx.x;               // query block
    const int g      = blockIdx.y;               // head-group (B*H)
    const size_t gbase = (size_t)g * LSEQ * DH;
    const float  scale = 0.08838834764831845f;   // 1/sqrt(128)

    // ---- Q strip -> A-layout f16 fragments, pre-scaled (stays in VGPRs) ----
    // A 16x32 f16 layout: lanes0-15 elems0-7 -> K 0..7, elems8-15 -> K 16..23;
    //                     lanes16-31 shift K by +8.
    v16h aq[4];
    {
        const int rowq = mb * BM + w * 16 + lanelo;
        const float* qp = q + gbase + (size_t)rowq * DH;
        #pragma unroll
        for (int c = 0; c < 4; ++c) {
            const float* p0 = qp + c * 32 + khalf * 8;
            v4f f0a = *(const v4f*)(p0);
            v4f f0b = *(const v4f*)(p0 + 4);
            v4f f1a = *(const v4f*)(p0 + 16);
            v4f f1b = *(const v4f*)(p0 + 20);
            #pragma unroll
            for (int e = 0; e < 4; ++e) {
                aq[c][e]      = (_Float16)(f0a[e] * scale);
                aq[c][4 + e]  = (_Float16)(f0b[e] * scale);
                aq[c][8 + e]  = (_Float16)(f1a[e] * scale);
                aq[c][12 + e] = (_Float16)(f1b[e] * scale);
            }
        }
    }

    v8f o[8];
    #pragma unroll
    for (int dt = 0; dt < 8; ++dt) o[dt] = {};
    float mstate[8], lstate[8];
    #pragma unroll
    for (int r = 0; r < 8; ++r) { mstate[r] = -__builtin_inff(); lstate[r] = 0.f; }

    #pragma unroll 1
    for (int nb = 0; nb < LSEQ / BN; ++nb) {
        __syncthreads();                          // previous reads done
        // ---- cooperative stage: K tile f16 row-major, V tile f16 transposed ----
        {
            const size_t blk = gbase + (size_t)nb * BN * DH;
            #pragma unroll
            for (int i = 0; i < 8; ++i) {
                const int idx = i * 256 + tid;            // float4 index in tile
                const int row = idx >> 5;                 // /32 float4 per row
                const int col = (idx & 31) * 4;
                v4f kf = *(const v4f*)(k + blk + (size_t)idx * 4);
                v4f vf = *(const v4f*)(v + blk + (size_t)idx * 4);
                _Float16* kd = &sK[row * SK_STR + col];
                #pragma unroll
                for (int j = 0; j < 4; ++j) kd[j] = (_Float16)kf[j];
                #pragma unroll
                for (int j = 0; j < 4; ++j) sVt[(col + j) * SV_STR + row] = (_Float16)vf[j];
                if (nb + 1 < LSEQ / BN) {                 // -> global_prefetch_b8
                    __builtin_prefetch(k + blk + BN * DH + (size_t)idx * 4, 0, 1);
                    __builtin_prefetch(v + blk + BN * DH + (size_t)idx * 4, 0, 1);
                }
            }
        }
        __syncthreads();

        // ---- S = Q K^T : 4 n-tiles x (Dh/32) WMMAs, fp32 accum ----
        v8f s[4];
        #pragma unroll
        for (int t = 0; t < 4; ++t) {
            v8f acc = {};
            #pragma unroll
            for (int c = 0; c < 4; ++c) {
                // B 32x16: elems0-7 -> k khalf*8+0..7, elems8-15 -> k 16+khalf*8+0..7
                const _Float16* bp = &sK[(t * 16 + lanelo) * SK_STR + c * 32 + khalf * 8];
                v8h lo = *(const v8h*)bp;
                v8h hi = *(const v8h*)(bp + 16);
                v16h bk;
                #pragma unroll
                for (int e = 0; e < 8; ++e) { bk[e] = lo[e]; bk[8 + e] = hi[e]; }
                acc = __builtin_amdgcn_wmma_f32_16x16x32_f16(
                          false, aq[c], false, bk, (short)0, acc, false, false);
            }
            s[t] = acc;
        }

        // ---- online softmax (fp32). Row r lives in slot r across a 16-lane half ----
        #pragma unroll
        for (int r = 0; r < 8; ++r) {
            float mx = fmaxf(fmaxf(s[0][r], s[1][r]), fmaxf(s[2][r], s[3][r]));
            #pragma unroll
            for (int off = 1; off < 16; off <<= 1) mx = fmaxf(mx, __shfl_xor(mx, off, 32));
            const float mnew  = fmaxf(mstate[r], mx);
            const float alpha = __expf(mstate[r] - mnew);
            lstate[r] *= alpha;
            #pragma unroll
            for (int dt = 0; dt < 8; ++dt) o[dt][r] *= alpha;
            float rsum = 0.f;
            #pragma unroll
            for (int t = 0; t < 4; ++t) {
                const float p = __expf(s[t][r] - mnew);
                s[t][r] = p;
                rsum += p;
            }
            #pragma unroll
            for (int off = 1; off < 16; off <<= 1) rsum += __shfl_xor(rsum, off, 32);
            lstate[r] += rsum;
            mstate[r]  = mnew;
        }

        // ---- P: C-layout -> per-wave LDS (row-major) -> A-layout fragments ----
        #pragma unroll
        for (int t = 0; t < 4; ++t)
            #pragma unroll
            for (int r = 0; r < 8; ++r)
                sP[w][(r + 8 * khalf) * SP_STR + t * 16 + lanelo] = (_Float16)s[t][r];
        asm volatile("s_wait_dscnt 0" ::: "memory");   // intra-wave cross-lane LDS dep

        v16h pa[2];
        #pragma unroll
        for (int kc = 0; kc < 2; ++kc) {
            const _Float16* pp = &sP[w][lanelo * SP_STR + kc * 32 + khalf * 8];
            v8h g0 = *(const v8h*)pp;          // K = khalf*8 + 0..7
            v8h g1 = *(const v8h*)(pp + 16);   // K = 16 + khalf*8 + 0..7
            v16h a;
            #pragma unroll
            for (int e = 0; e < 8; ++e) { a[e] = g0[e]; a[8 + e] = g1[e]; }
            pa[kc] = a;
        }

        // ---- O += P V : 8 d-tiles x 2 k-chunks WMMAs ----
        #pragma unroll
        for (int dt = 0; dt < 8; ++dt) {
            #pragma unroll
            for (int kc = 0; kc < 2; ++kc) {
                const _Float16* vp = &sVt[(dt * 16 + lanelo) * SV_STR + kc * 32 + khalf * 8];
                v8h lo = *(const v8h*)vp;
                v8h hi = *(const v8h*)(vp + 16);
                v16h bv;
                #pragma unroll
                for (int e = 0; e < 8; ++e) { bv[e] = lo[e]; bv[8 + e] = hi[e]; }
                o[dt] = __builtin_amdgcn_wmma_f32_16x16x32_f16(
                            false, pa[kc], false, bv, (short)0, o[dt], false, false);
            }
        }
    }

    // ---- epilogue: O / l, scatter to [B, L, H, Dh] ----
    const int b = g >> 4, h = g & (NHEADS - 1);
    #pragma unroll
    for (int r = 0; r < 8; ++r) {
        const int   rowg = mb * BM + w * 16 + r + 8 * khalf;
        const float inv  = 1.f / lstate[r];
        const size_t base = (((size_t)b * LSEQ + rowg) * NHEADS + h) * DH;
        #pragma unroll
        for (int dt = 0; dt < 8; ++dt)
            out[base + dt * 16 + lanelo] = o[dt][r] * inv;
    }
}

extern "C" void kernel_launch(void* const* d_in, const int* in_sizes, int n_in,
                              void* d_out, int out_size, void* d_ws, size_t ws_size,
                              hipStream_t stream) {
    const float* q = (const float*)d_in[0];
    const float* k = (const float*)d_in[1];
    const float* v = (const float*)d_in[2];
    float* out = (float*)d_out;
    (void)in_sizes; (void)n_in; (void)out_size; (void)d_ws; (void)ws_size;
    dim3 grid(LSEQ / BM, GROUPS);   // (16, 32)
    dim3 block(256);                 // 8 wave32 waves
    fa_fwd_f32_wmma<<<grid, block, 0, stream>>>(q, k, v, out);
}
